// MultiHeadAttention_26860725469573
// MI455X (gfx1250) — matrix operand order repaired and checked by reading, not compiled, not run on hardware
//
#include <hip/hip_runtime.h>

// ---------------------------------------------------------------------------
// MI455X (gfx1250) multi-head attention forward.
// B=2, S=2048, D=1024, H=16, HD=64. f32 in/out, bf16 WMMA internally.
// ---------------------------------------------------------------------------

typedef __attribute__((ext_vector_type(16))) __bf16         v16bf;
typedef __attribute__((ext_vector_type(8)))  float          v8f;
typedef __attribute__((ext_vector_type(4)))  float          fx4;
typedef __attribute__((ext_vector_type(8)))  unsigned short u16x8;

union Frag {
  v16bf v;
  fx4   f4[2];
  u16x8 u8[2];
};

constexpr int Bsz = 2;
constexpr int Seq = 2048;
constexpr int Dm  = 1024;
constexpr int Hn  = 16;
constexpr int Hd  = 64;
constexpr int GM  = Bsz * Seq;   // 4096 rows (tokens)
constexpr int GN  = Dm;          // 1024
constexpr int GK  = Dm;          // 1024

// RNE f32 -> bf16 (no NaN path: all converted values are finite or exact 0)
__device__ __forceinline__ unsigned short f32_to_bf16(float f) {
  union { float f; unsigned int u; } cv; cv.f = f;
  unsigned int u = cv.u;
  return (unsigned short)((u + 0x7fffu + ((u >> 16) & 1u)) >> 16);
}

__device__ __forceinline__ v8f wmma_bf16(v16bf a, v16bf b, v8f c) {
  // D = A(16x32 bf16) * B(32x16 bf16) + C(16x16 f32)
  return __builtin_amdgcn_wmma_f32_16x16x32_bf16(
      /*neg_a=*/false, a, /*neg_b=*/false, b,
      /*c_mod=*/(short)0, c, /*reuse_a=*/false, /*reuse_b=*/false);
}

__device__ __forceinline__ float rowmax16(float v) {
  v = fmaxf(v, __shfl_xor(v, 1, 32));
  v = fmaxf(v, __shfl_xor(v, 2, 32));
  v = fmaxf(v, __shfl_xor(v, 4, 32));
  v = fmaxf(v, __shfl_xor(v, 8, 32));
  return v;
}
__device__ __forceinline__ float rowsum16(float v) {
  v += __shfl_xor(v, 1, 32);
  v += __shfl_xor(v, 2, 32);
  v += __shfl_xor(v, 4, 32);
  v += __shfl_xor(v, 8, 32);
  return v;
}

// ---------------------------------------------------------------------------
// Weight convert + transpose: W[K=1024][N=1024] f32 -> Wt[N][K] bf16.
// Contraction dim contiguous per output column => B-fragment loads are b128.
// ---------------------------------------------------------------------------
__global__ __launch_bounds__(256) void convert_wt_kernel(
    const float* __restrict__ W, unsigned short* __restrict__ Wt) {
  size_t g = (size_t)blockIdx.x * blockDim.x + threadIdx.x;  // 1M threads
  int n = (int)(g >> 10);
  int k = (int)(g & 1023);
  Wt[(size_t)n * GK + k] = f32_to_bf16(W[(size_t)k * GN + n]);
}

// ---------------------------------------------------------------------------
// bf16 WMMA GEMM: C[M,N] = A[M,K](f32) * Wt[N,K](bf16)^T + bias.
//   MODE 0: store bf16, scatter to [B,H,S,HD]          (Q, K projections)
//   MODE 1: store bf16, scatter to [B,H,HD,S]          (V, pre-transposed)
//   MODE 2: store f32 row-major [M,N]                  (output projection)
// Block: 256 threads (8 waves), tile 128x128, K-step 32.
// Wave grid 4(M) x 2(N); each wave owns 2x4 = 8 WMMA accumulators.
// ---------------------------------------------------------------------------
template <int MODE>
__global__ __launch_bounds__(256, 1) void gemm128_kernel(
    const float* __restrict__ A, const unsigned short* __restrict__ Bt,
    const float* __restrict__ bias, void* __restrict__ outp) {
  __shared__ __align__(16) unsigned short As[128 * 32];
  __shared__ __align__(16) unsigned short Bs[128 * 32];

  const int tid   = threadIdx.x;
  const int lane  = tid & 31;
  const int wave  = tid >> 5;
  const int waveM = wave & 3;        // 0..3 -> 32-row strips
  const int waveN = wave >> 2;       // 0..1 -> 64-col strips
  const int hlf   = lane >> 4;       // half-wave select
  const int l15   = lane & 15;
  const int rowBase = blockIdx.y * 128;
  const int colBase = blockIdx.x * 128;

  v8f acc[2][4];
#pragma unroll
  for (int i = 0; i < 2; ++i)
#pragma unroll
    for (int j = 0; j < 4; ++j) acc[i][j] = 0.f;

  for (int k0 = 0; k0 < GK; k0 += 32) {
    // ---- stage A tile (128x32 f32 -> bf16 in LDS), 2 chunks of 8 floats ----
#pragma unroll
    for (int i = 0; i < 2; ++i) {
      int c    = tid + i * 256;        // 0..511
      int row  = c >> 2;               // 0..127
      int koff = (c & 3) * 8;          // 0,8,16,24
      const float* src = A + (size_t)(rowBase + row) * GK + k0 + koff;
      fx4 lo = *(const fx4*)src;
      fx4 hi = *(const fx4*)(src + 4);
      u16x8 pk;
      pk[0] = f32_to_bf16(lo[0]); pk[1] = f32_to_bf16(lo[1]);
      pk[2] = f32_to_bf16(lo[2]); pk[3] = f32_to_bf16(lo[3]);
      pk[4] = f32_to_bf16(hi[0]); pk[5] = f32_to_bf16(hi[1]);
      pk[6] = f32_to_bf16(hi[2]); pk[7] = f32_to_bf16(hi[3]);
      *(u16x8*)&As[row * 32 + koff] = pk;
    }
    // ---- stage B tile (128 cols x 32 k, bf16), 2 chunks of 16B ----
#pragma unroll
    for (int i = 0; i < 2; ++i) {
      int c    = tid + i * 256;
      int n    = c >> 2;
      int koff = (c & 3) * 8;
      const unsigned short* src = Bt + (size_t)(colBase + n) * GK + k0 + koff;
      *(u16x8*)&Bs[n * 32 + koff] = *(const u16x8*)src;
    }
    __syncthreads();

    // ---- fragments (ISA VGPR striping for 16-bit A 16x32 / B 32x16) ----
    Frag af[2], bf[4];
#pragma unroll
    for (int mt = 0; mt < 2; ++mt) {
      int r = (waveM * 32 + mt * 16 + l15) * 32;
      af[mt].f4[0] = *(const fx4*)&As[r + hlf * 8];        // K = h*8 + 0..7
      af[mt].f4[1] = *(const fx4*)&As[r + 16 + hlf * 8];   // K = 16+h*8+0..7
    }
#pragma unroll
    for (int nt = 0; nt < 4; ++nt) {
      int r = (waveN * 64 + nt * 16 + l15) * 32;
      bf[nt].f4[0] = *(const fx4*)&Bs[r + hlf * 8];        // K = h*8 + 0..7
      bf[nt].f4[1] = *(const fx4*)&Bs[r + 16 + hlf * 8];   // K = 16+h*8+0..7
    }
#pragma unroll
    for (int mt = 0; mt < 2; ++mt)
#pragma unroll
      for (int nt = 0; nt < 4; ++nt)
        acc[mt][nt] = wmma_bf16(af[mt].v, bf[nt].v, acc[mt][nt]);
    __syncthreads();
  }

  // ---- epilogue ----
#pragma unroll
  for (int mt = 0; mt < 2; ++mt) {
#pragma unroll
    for (int nt = 0; nt < 4; ++nt) {
      int col = colBase + waveN * 64 + nt * 16 + l15;
      float bv = bias[col];
#pragma unroll
      for (int r = 0; r < 8; ++r) {
        int row   = rowBase + waveM * 32 + mt * 16 + hlf * 8 + r;
        float val = acc[mt][nt][r] + bv;
        if constexpr (MODE == 2) {
          ((float*)outp)[(size_t)row * GN + col] = val;
        } else {
          int b = row >> 11, s = row & (Seq - 1);
          int h = col >> 6,  d = col & (Hd - 1);
          size_t idx;
          if constexpr (MODE == 0)        // [B,H,S,HD]
            idx = (((size_t)b * Hn + h) * Seq + s) * Hd + d;
          else                            // [B,H,HD,S]
            idx = (((size_t)b * Hn + h) * Hd + d) * Seq + s;
          ((unsigned short*)outp)[idx] = f32_to_bf16(val);
        }
      }
    }
  }
}

// ---------------------------------------------------------------------------
// Flash attention: grid (S/64, B*H), block 128 (4 waves).
// Each wave owns one 16-row Q tile; key blocks of 64; online softmax.
// The 64-key K/V tiles are shared by all 4 waves -> staged once per block
// into LDS (coalesced global_load_b128), fragments read via ds_load_b128.
//   qh [BH][S][HD] bf16, kh [BH][S][HD] bf16, vt [BH][HD][S] bf16
//   ao [B][S][D] f32  (attention output, reshaped for output projection)
// ---------------------------------------------------------------------------
constexpr int LP = 72;  // padded LDS row stride (bf16 elems) -> conflict-free

__global__ __launch_bounds__(128, 1) void attention_kernel(
    const unsigned short* __restrict__ qh, const unsigned short* __restrict__ kh,
    const unsigned short* __restrict__ vt, float* __restrict__ ao) {
  __shared__ __align__(16) unsigned short P[4][16][LP];  // per-wave P tile
  __shared__ __align__(16) unsigned short Ks[64 * LP];   // [key][HD]
  __shared__ __align__(16) unsigned short Vs[64 * LP];   // [HD][key]

  const int tid  = threadIdx.x;
  const int lane = tid & 31;
  const int wave = tid >> 5;
  const int hlf  = lane >> 4;
  const int l15  = lane & 15;
  const int bh   = blockIdx.y;
  const int b    = bh >> 4;
  const int h    = bh & (Hn - 1);
  const int blockBase = blockIdx.x * 64;
  const int qbase     = blockBase + wave * 16;

  // ---- load Q fragments once (A layout: row=l15, K striped by half) ----
  Frag qa[2];
  const unsigned short* qp = qh + ((size_t)bh * Seq + (qbase + l15)) * Hd;
#pragma unroll
  for (int f = 0; f < 2; ++f) {
    qa[f].f4[0] = *(const fx4*)(qp + f * 32 + hlf * 8);
    qa[f].f4[1] = *(const fx4*)(qp + f * 32 + 16 + hlf * 8);
  }

  v8f o[4];
#pragma unroll
  for (int t = 0; t < 4; ++t) o[t] = 0.f;
  float mrow[8], lrow[8];
#pragma unroll
  for (int r = 0; r < 8; ++r) { mrow[r] = -__builtin_inff(); lrow[r] = 0.f; }

  const float scale = 0.125f;  // 1/sqrt(64)

  // Uniform trip count for all waves in the block (barrier-safe).
  for (int j0 = 0; j0 < blockBase + 64; j0 += 64) {
    // ---- cooperative stage: K tile [64 keys][64 HD], V tile [64 HD][64 keys]
    // 512 chunks of 8 bf16 (16B) each; 128 threads -> 4 chunks per tile.
#pragma unroll
    for (int i = 0; i < 4; ++i) {
      int c    = tid + i * 128;        // 0..511
      int row  = c >> 3;               // 0..63
      int eoff = (c & 7) * 8;          // 0..56
      // K: row = key position, eoff = HD offset
      *(u16x8*)&Ks[row * LP + eoff] =
          *(const u16x8*)(kh + ((size_t)bh * Seq + j0 + row) * Hd + eoff);
      // V: row = HD index, eoff = key offset (vt is [BH][HD][S])
      *(u16x8*)&Vs[row * LP + eoff] =
          *(const u16x8*)(vt + ((size_t)bh * Hd + row) * Seq + j0 + eoff);
    }
    if (j0 + 64 < Seq)  // pull next K block toward the WGP (global_prefetch_b8)
      __builtin_prefetch(kh + ((size_t)bh * Seq + j0 + 64 + (tid >> 1)) * Hd, 0, 0);
    __syncthreads();

    // ---- scores S = Q K^T for a 16x64 tile (4 N-tiles, HD chained 2x32) ----
    v8f sc[4];
#pragma unroll
    for (int nt = 0; nt < 4; ++nt) {
      int key = nt * 16 + l15;  // B column = key position (block-local)
      Frag kb0, kb1;
      kb0.f4[0] = *(const fx4*)&Ks[key * LP + hlf * 8];
      kb0.f4[1] = *(const fx4*)&Ks[key * LP + 16 + hlf * 8];
      kb1.f4[0] = *(const fx4*)&Ks[key * LP + 32 + hlf * 8];
      kb1.f4[1] = *(const fx4*)&Ks[key * LP + 32 + 16 + hlf * 8];
      v8f z = 0.f;
      v8f s0 = wmma_bf16(qa[0].v, kb0.v, z);
      sc[nt]  = wmma_bf16(qa[1].v, kb1.v, s0);
      // scale + causal mask
#pragma unroll
      for (int r = 0; r < 8; ++r) {
        int qrow = qbase + hlf * 8 + r;
        int kpos = j0 + nt * 16 + l15;
        float v  = sc[nt][r] * scale;
        sc[nt][r] = (kpos <= qrow) ? v : -__builtin_inff();
      }
    }

    // ---- online softmax statistics ----
    float corr[8];
#pragma unroll
    for (int r = 0; r < 8; ++r) {
      float tm = fmaxf(fmaxf(sc[0][r], sc[1][r]), fmaxf(sc[2][r], sc[3][r]));
      tm = rowmax16(tm);
      float mnew = fmaxf(mrow[r], tm);
      corr[r] = __expf(mrow[r] - mnew);
      mrow[r] = mnew;
    }
#pragma unroll
    for (int r = 0; r < 8; ++r) {
      float ts = 0.f;
#pragma unroll
      for (int nt = 0; nt < 4; ++nt) {
        float p = __expf(sc[nt][r] - mrow[r]);
        sc[nt][r] = p;
        ts += p;
      }
      ts = rowsum16(ts);
      lrow[r] = lrow[r] * corr[r] + ts;
    }
#pragma unroll
    for (int t = 0; t < 4; ++t)
#pragma unroll
      for (int r = 0; r < 8; ++r) o[t][r] *= corr[r];

    // ---- re-stripe P: C layout -> A-fragment layout via LDS ----
#pragma unroll
    for (int nt = 0; nt < 4; ++nt)
#pragma unroll
      for (int r = 0; r < 8; ++r)
        P[wave][hlf * 8 + r][nt * 16 + l15] = f32_to_bf16(sc[nt][r]);
    __syncthreads();

    Frag pa[2];
#pragma unroll
    for (int f = 0; f < 2; ++f) {
      pa[f].f4[0] = *(const fx4*)&P[wave][l15][f * 32 + hlf * 8];
      pa[f].f4[1] = *(const fx4*)&P[wave][l15][f * 32 + 16 + hlf * 8];
    }

    // ---- O += P V  (V tile in LDS: row = HD index, K = key pos) ----
#pragma unroll
    for (int ont = 0; ont < 4; ++ont) {
      int d = ont * 16 + l15;  // B column = HD index
      Frag vb0, vb1;
      vb0.f4[0] = *(const fx4*)&Vs[d * LP + hlf * 8];
      vb0.f4[1] = *(const fx4*)&Vs[d * LP + 16 + hlf * 8];
      vb1.f4[0] = *(const fx4*)&Vs[d * LP + 32 + hlf * 8];
      vb1.f4[1] = *(const fx4*)&Vs[d * LP + 32 + 16 + hlf * 8];
      o[ont] = wmma_bf16(pa[0].v, vb0.v, o[ont]);
      o[ont] = wmma_bf16(pa[1].v, vb1.v, o[ont]);
    }
    // WAR guard: next iteration re-stages Ks/Vs and rewrites P.
    __syncthreads();
  }

  // ---- normalize and write [B,S,D] f32 ----
  float inv[8];
#pragma unroll
  for (int r = 0; r < 8; ++r) inv[r] = 1.0f / lrow[r];
#pragma unroll
  for (int ont = 0; ont < 4; ++ont)
#pragma unroll
    for (int r = 0; r < 8; ++r) {
      int srow = qbase + hlf * 8 + r;
      ao[((size_t)b * Seq + srow) * Dm + h * Hd + ont * 16 + l15] =
          o[ont][r] * inv[r];
    }
}

// ---------------------------------------------------------------------------
// Launch. Inputs: x, attn_mask, Wq, bq, Wk, bk, Wv, bv, Wo, bo (f32).
// Workspace: Wt x4 (8 MiB) + qh/kh/vt bf16 (24 MiB) + ao f32 (16 MiB) = 48 MiB.
// ---------------------------------------------------------------------------
extern "C" void kernel_launch(void* const* d_in, const int* in_sizes, int n_in,
                              void* d_out, int out_size, void* d_ws, size_t ws_size,
                              hipStream_t stream) {
  const float* x  = (const float*)d_in[0];
  // d_in[1] = attn_mask (causal, handled analytically)
  const float* Wq = (const float*)d_in[2];
  const float* bq = (const float*)d_in[3];
  const float* Wk = (const float*)d_in[4];
  const float* bk = (const float*)d_in[5];
  const float* Wv = (const float*)d_in[6];
  const float* bv = (const float*)d_in[7];
  const float* Wo = (const float*)d_in[8];
  const float* bo = (const float*)d_in[9];

  unsigned short* ws16 = (unsigned short*)d_ws;
  size_t off = 0;
  unsigned short* Wqb = ws16 + off; off += (size_t)GK * GN;
  unsigned short* Wkb = ws16 + off; off += (size_t)GK * GN;
  unsigned short* Wvb = ws16 + off; off += (size_t)GK * GN;
  unsigned short* Wob = ws16 + off; off += (size_t)GK * GN;
  unsigned short* qhb = ws16 + off; off += (size_t)GM * GN;
  unsigned short* khb = ws16 + off; off += (size_t)GM * GN;
  unsigned short* vtb = ws16 + off; off += (size_t)GM * GN;
  float* ao = (float*)(ws16 + off);   // 16B-aligned (off*2 is a multiple of 16)

  // 1) weight transpose+convert
  convert_wt_kernel<<<4096, 256, 0, stream>>>(Wq, Wqb);
  convert_wt_kernel<<<4096, 256, 0, stream>>>(Wk, Wkb);
  convert_wt_kernel<<<4096, 256, 0, stream>>>(Wv, Wvb);
  convert_wt_kernel<<<4096, 256, 0, stream>>>(Wo, Wob);

  // 2) QKV projections (WMMA GEMMs), scattered into attention layouts
  dim3 gg(GN / 128, GM / 128);  // (8, 32)
  gemm128_kernel<0><<<gg, 256, 0, stream>>>(x, Wqb, bq, qhb);
  gemm128_kernel<0><<<gg, 256, 0, stream>>>(x, Wkb, bk, khb);
  gemm128_kernel<1><<<gg, 256, 0, stream>>>(x, Wvb, bv, vtb);

  // 3) causal flash attention (K/V staged in LDS, shared by 4 waves)
  attention_kernel<<<dim3(Seq / 64, Bsz * Hn), 128, 0, stream>>>(qhb, khb, vtb, ao);

  // 4) output projection -> d_out (f32)
  gemm128_kernel<2><<<gg, 256, 0, stream>>>(ao, Wob, bo, (float*)d_out);
}
